// PaGNNEncoder_51101520888246
// MI455X (gfx1250) — matrix operand order repaired and checked by reading, not compiled, not run on hardware
//
#include <hip/hip_runtime.h>

// ---------------------------------------------------------------------------
// PaGNN encoder forward for MI455X (gfx1250, wave32).
// Dense matmuls use v_wmma_f32_16x16x32_bf16 with a bf16x3 split (hi*hi +
// hi*lo + lo*hi) for ~fp32 accuracy; GEMMs here are HBM/L2 bound (<26MB
// activations each vs 23.3 TB/s), so the 3x WMMA cost is free.
// Edge softmax/scatter stays fp32 with atomics, pruned exactly to the
// broadcast frontier / aggregation target set.
// ---------------------------------------------------------------------------

typedef __attribute__((ext_vector_type(16))) __bf16 v16bf;
typedef __attribute__((ext_vector_type(8)))  float  v8f;

static __device__ __forceinline__ float lrelu(float x) { return x > 0.0f ? x : 0.2f * x; }
static __device__ __forceinline__ unsigned fenc(float f) {
  unsigned u = __float_as_uint(f);
  return (u & 0x80000000u) ? ~u : (u | 0x80000000u);
}
static __device__ __forceinline__ float fdec(unsigned u) {
  unsigned v = (u & 0x80000000u) ? (u & 0x7fffffffu) : ~u;
  return __uint_as_float(v);
}
static __device__ __forceinline__ float sigm(float x) { return 1.0f / (1.0f + expf(-x)); }

// ------------------------------ fill kernels -------------------------------
__global__ void k_fill_f32(float* p, float v, int count) {
  int i = blockIdx.x * blockDim.x + threadIdx.x;
  if (i < count) p[i] = v;
}
__global__ void k_fill_u32(unsigned* p, unsigned v, int count) {
  int i = blockIdx.x * blockDim.x + threadIdx.x;
  if (i < count) p[i] = v;
}

// ------------------------------ WMMA GEMM ----------------------------------
// Y(nrows x dout) = X(nrows x K) @ W(dout x K)^T [+ bias] [+ Y (accumulate)]
// X row k<Ka comes from Xa, k>=Ka from Xb (concat support). K,Ka multiples of 32.
// One wave computes one 16x16 output tile via v_wmma_f32_16x16x32_bf16.
__global__ __launch_bounds__(256) void k_gemm_bf16x3(
    const float* __restrict__ Xa, const float* __restrict__ Xb, int Ka, int K,
    const float* __restrict__ W, const float* __restrict__ bias,
    float* __restrict__ Y, int nrows, int dout, int accumulate)
{
  const int gwave = (int)((blockIdx.x * blockDim.x + threadIdx.x) >> 5);
  const int lane  = threadIdx.x & 31;
  const int tilesN = dout >> 4;
  const int total  = (nrows >> 4) * tilesN;
  if (gwave >= total) return;
  const int row0 = (gwave / tilesN) << 4;
  const int col0 = (gwave % tilesN) << 4;
  const int hlf = lane >> 4;   // half-wave: selects K sub-group per ISA layout
  const int mr  = lane & 15;   // row (A) / output column (B,C)

  v8f acc = {0.0f, 0.0f, 0.0f, 0.0f, 0.0f, 0.0f, 0.0f, 0.0f};

  for (int k0 = 0; k0 < K; k0 += 32) {
    const float* xp; int ldx, kb;
    if (k0 < Ka) { xp = Xa; ldx = Ka;     kb = k0; }
    else         { xp = Xb; ldx = K - Ka; kb = k0 - Ka; }
    const float* xr = xp + (size_t)(row0 + mr) * (size_t)ldx + kb;

    // A fragment (16x32 bf16): element e<8 -> K = 8*hlf + e ; e>=8 -> K = 16 + 8*hlf + (e-8)
    float4 a0 = *(const float4*)(xr + 8 * hlf);
    float4 a1 = *(const float4*)(xr + 8 * hlf + 4);
    float4 a2 = *(const float4*)(xr + 16 + 8 * hlf);
    float4 a3 = *(const float4*)(xr + 16 + 8 * hlf + 4);
    // B fragment (32x16): B[k][n] = W[col0+n][k]; element e<8 -> K = 8*hlf + e ; e>=8 -> K = 16 + 8*hlf + (e-8)
    const float* wr = W + (size_t)(col0 + mr) * (size_t)K + k0 + 8 * hlf;
    float4 b0 = *(const float4*)(wr);
    float4 b1 = *(const float4*)(wr + 4);
    float4 b2 = *(const float4*)(wr + 16);
    float4 b3 = *(const float4*)(wr + 20);

    float af[16]  = {a0.x,a0.y,a0.z,a0.w, a1.x,a1.y,a1.z,a1.w,
                     a2.x,a2.y,a2.z,a2.w, a3.x,a3.y,a3.z,a3.w};
    float bfv[16] = {b0.x,b0.y,b0.z,b0.w, b1.x,b1.y,b1.z,b1.w,
                     b2.x,b2.y,b2.z,b2.w, b3.x,b3.y,b3.z,b3.w};

    v16bf Ah, Al, Bh, Bl;
#pragma unroll
    for (int e = 0; e < 16; ++e) {
      __bf16 ah = (__bf16)af[e];
      __bf16 bh = (__bf16)bfv[e];
      Ah[e] = ah; Al[e] = (__bf16)(af[e]  - (float)ah);
      Bh[e] = bh; Bl[e] = (__bf16)(bfv[e] - (float)bh);
    }
    // bf16x3: (Ah+Al)(Bh+Bl) ~= Al*Bh + Ah*Bl + Ah*Bh  (drop lo*lo)
    acc = __builtin_amdgcn_wmma_f32_16x16x32_bf16(false, Al, false, Bh, (short)0, acc, false, false);
    acc = __builtin_amdgcn_wmma_f32_16x16x32_bf16(false, Ah, false, Bl, (short)0, acc, false, false);
    acc = __builtin_amdgcn_wmma_f32_16x16x32_bf16(false, Ah, false, Bh, (short)0, acc, false, false);
  }

  // Epilogue. C layout: VGPR r: M = r + 8*hlf, N = mr.  bias[nn] is invariant
  // across the 8 accumulator rows -> load once; keep accumulate as a uniform
  // branch so each path is straight-line.
  const int nn = col0 + mr;
  const float bv = bias ? bias[nn] : 0.0f;
  float* yr = Y + (size_t)(row0 + (hlf << 3)) * (size_t)dout + nn;
  if (accumulate) {
#pragma unroll
    for (int r = 0; r < 8; ++r) {
      float* yp = yr + (size_t)r * (size_t)dout;
      *yp = *yp + acc[r] + bv;
    }
  } else {
#pragma unroll
    for (int r = 0; r < 8; ++r) {
      yr[(size_t)r * (size_t)dout] = acc[r] + bv;
    }
  }
}

// --------------------------- attention dot (wave/node) ---------------------
__global__ void k_attdot(const float* __restrict__ h, const float* __restrict__ att_src,
                         const float* __restrict__ att_dst, float* __restrict__ asrc,
                         float* __restrict__ adst, int n)
{
  int w = (int)((blockIdx.x * blockDim.x + threadIdx.x) >> 5);
  int lane = threadIdx.x & 31;
  if (w >= n) return;
  const float* hr = h + (size_t)w * 64;
  float h0 = hr[lane], h1 = hr[lane + 32];
  float s = h0 * att_src[lane] + h1 * att_src[lane + 32];
  float d = h0 * att_dst[lane] + h1 * att_dst[lane + 32];
#pragma unroll
  for (int off = 16; off > 0; off >>= 1) {   // wave32 reduction
    s += __shfl_xor(s, off, 32);
    d += __shfl_xor(d, off, 32);
  }
  if (lane == 0) { asrc[w] = s; adst[w] = d; }
}

// ------------------------------ edge kernels -------------------------------
__global__ void k_edge_max(const int* __restrict__ row, const int* __restrict__ col,
                           const float* __restrict__ asrc, const float* __restrict__ adst,
                           const int* __restrict__ vmask, const int* __restrict__ tmask,
                           unsigned* __restrict__ menc, int E)
{
  int i = blockIdx.x * blockDim.x + threadIdx.x;
  if (i >= E) return;
  int r = row[i], c = col[i];
  if (vmask && !vmask[r]) return;
  if (tmask && !tmask[c]) return;
  float e = lrelu(asrc[r] + adst[c]);
  atomicMax(&menc[c], fenc(e));
}

__global__ void k_node_m(const float* __restrict__ asrc, const float* __restrict__ adst,
                         unsigned* __restrict__ menc, float* __restrict__ wself,
                         float* __restrict__ denom, int n)
{
  int i = blockIdx.x * blockDim.x + threadIdx.x;
  if (i >= n) return;
  float es = lrelu(asrc[i] + adst[i]);
  float mm = fmaxf(fdec(menc[i]), es);
  menc[i] = __float_as_uint(mm);      // m now stored as plain float bits
  wself[i] = expf(es - mm);
  denom[i] = 0.0f;
}

__global__ void k_edge_w(const int* __restrict__ row, const int* __restrict__ col,
                         const float* __restrict__ asrc, const float* __restrict__ adst,
                         const int* __restrict__ vmask, const int* __restrict__ tmask,
                         const float* __restrict__ m, float* __restrict__ wbuf,
                         float* __restrict__ denom, int E)
{
  int i = blockIdx.x * blockDim.x + threadIdx.x;
  if (i >= E) return;
  int r = row[i], c = col[i];
  float w = 0.0f;
  if ((!vmask || vmask[r]) && (!tmask || tmask[c])) {
    float e = lrelu(asrc[r] + adst[c]);
    w = expf(e - m[c]);
    atomicAdd(&denom[c], w);
  }
  wbuf[i] = w;
}

__global__ void k_edge_num(const int* __restrict__ row, const int* __restrict__ col,
                           const float* __restrict__ wbuf, const float* __restrict__ hg,
                           float* __restrict__ num, int total /* = E*64 */)
{
  int idx = blockIdx.x * blockDim.x + threadIdx.x;
  if (idx >= total) return;
  int e = idx >> 6, d = idx & 63;
  float w = wbuf[e];
  if (w == 0.0f) return;
  atomicAdd(&num[(size_t)col[e] * 64 + d], w * hg[(size_t)row[e] * 64 + d]);
}

__global__ void k_gat_final(const float* __restrict__ num, const float* __restrict__ hg,
                            const float* __restrict__ wself, const float* __restrict__ denom,
                            const float* __restrict__ bias, float* __restrict__ out,
                            int total /* = n*64 */, int do_tanh)
{
  int idx = blockIdx.x * blockDim.x + threadIdx.x;
  if (idx >= total) return;
  int i = idx >> 6, d = idx & 63;
  float ws = wself[i];
  float v = (num[idx] + ws * hg[idx]) / (denom[i] + ws) + bias[d];
  out[idx] = do_tanh ? tanhf(v) : v;
}

// ------------------------------ LSTM / misc --------------------------------
__global__ void k_lstm(const float* __restrict__ g, float* __restrict__ h,
                       float* __restrict__ c, int total /* = n*64 */)
{
  int idx = blockIdx.x * blockDim.x + threadIdx.x;
  if (idx >= total) return;
  int i = idx >> 6, d = idx & 63;
  const float* gr = g + (size_t)i * 256;
  float vi = gr[d], vf = gr[64 + d], vg = gr[128 + d], vo = gr[192 + d];
  float cc = sigm(vf) * c[idx] + sigm(vi) * tanhf(vg);
  c[idx] = cc;
  h[idx] = sigm(vo) * tanhf(cc);
}

__global__ void k_set_roots(int* mask, const int* roots, int off, int stride, int count) {
  int i = blockIdx.x * blockDim.x + threadIdx.x;
  if (i < count) mask[roots[off + i * stride]] = 1;
}

__global__ void k_touched(const int* __restrict__ row, const int* __restrict__ col,
                          const int* __restrict__ mask, int* __restrict__ touched, int E)
{
  int i = blockIdx.x * blockDim.x + threadIdx.x;
  if (i >= E) return;
  int r = row[i];
  if (mask[r]) { touched[r] = 1; touched[col[i]] = 1; }
}

__global__ void k_init_x(float* __restrict__ x, const float* __restrict__ feat,
                         const int* __restrict__ mask, int total)
{
  int idx = blockIdx.x * blockDim.x + threadIdx.x;
  if (idx >= total) return;
  x[idx] = mask[idx >> 6] ? feat[idx] : 0.0f;
}

__global__ void k_update_x(float* __restrict__ x, const float* __restrict__ h,
                           const float* __restrict__ feat, const int* __restrict__ touched, int total)
{
  int idx = blockIdx.x * blockDim.x + threadIdx.x;
  if (idx >= total) return;
  if (touched[idx >> 6]) x[idx] = h[idx] + feat[idx];
}

__global__ void k_gather(float* __restrict__ out, const float* __restrict__ src,
                         const int* __restrict__ roots, int off, int stride,
                         int colOff, int count)
{
  int idx = blockIdx.x * blockDim.x + threadIdx.x;
  if (idx >= count * 64) return;
  int i = idx >> 6, d = idx & 63;
  out[(size_t)i * 128 + colOff + d] = src[(size_t)roots[off + i * stride] * 64 + d];
}

// ============================ host orchestration ===========================

static inline int cdiv(long long a, int b) { return (int)((a + b - 1) / b); }

struct GatP  { const float *att_dst, *att_src, *bias, *w; };
struct LstmP { const float *bhh, *bih, *whh, *wih; };
struct AggP  { GatP gat[2]; const float *lin1_b, *lin1_w, *lin2_b, *lin2_w; LstmP lstm[2]; };
struct BroP  { GatP gat[2]; LstmP lstm[2]; };

static GatP take_gat(void* const* d, int& i) {
  GatP g; g.att_dst = (const float*)d[i++]; g.att_src = (const float*)d[i++];
  g.bias = (const float*)d[i++]; g.w = (const float*)d[i++]; return g;
}
static LstmP take_lstm(void* const* d, int& i) {
  LstmP l; l.bhh = (const float*)d[i++]; l.bih = (const float*)d[i++];
  l.whh = (const float*)d[i++]; l.wih = (const float*)d[i++]; return l;
}
static AggP take_agg(void* const* d, int& i) {
  AggP a; a.gat[0] = take_gat(d, i); a.gat[1] = take_gat(d, i);
  a.lin1_b = (const float*)d[i++]; a.lin1_w = (const float*)d[i++];
  a.lin2_b = (const float*)d[i++]; a.lin2_w = (const float*)d[i++];
  a.lstm[0] = take_lstm(d, i); a.lstm[1] = take_lstm(d, i); return a;
}
static BroP take_bro(void* const* d, int& i) {
  BroP b; b.gat[0] = take_gat(d, i); b.lstm[0] = take_lstm(d, i);
  b.gat[1] = take_gat(d, i); b.lstm[1] = take_lstm(d, i); return b;
}

struct Ctx {
  hipStream_t s;
  int n, E, Ebi;
  const float* feat;
  const int *erow, *ecol, *brow, *bcol, *roots;
  float *bx, *bh, *bc, *bhg, *bnum, *bt1, *bt2, *bx0, *bg;
  float *m, *wself, *denom, *asrc, *adst, *wbuf;
  int *maskA, *maskB, *tmask;
  float* out;
};

static void gemm(const Ctx& C, const float* Xa, const float* Xb, int Ka, int K,
                 const float* W, const float* bias, float* Y, int dout, int accumulate)
{
  int tiles = (C.n >> 4) * (dout >> 4);     // n = 50000 is a multiple of 16
  k_gemm_bf16x3<<<cdiv(tiles, 8), 256, 0, C.s>>>(Xa, Xb, Ka, K, W, bias, Y, C.n, dout, accumulate);
}

static void run_gat(const Ctx& C, const GatP& g, const float* x,
                    const int* row, const int* col, int E,
                    const int* vmask, const int* tmask, float* outb, int do_tanh)
{
  gemm(C, x, nullptr, 64, 64, g.w, nullptr, C.bhg, 64, 0);
  k_attdot<<<cdiv(C.n, 8), 256, 0, C.s>>>(C.bhg, g.att_src, g.att_dst, C.asrc, C.adst, C.n);
  k_fill_u32<<<cdiv(C.n, 256), 256, 0, C.s>>>((unsigned*)C.m, 0x007FFFFFu, C.n); // enc(-inf)
  k_edge_max<<<cdiv(E, 256), 256, 0, C.s>>>(row, col, C.asrc, C.adst, vmask, tmask, (unsigned*)C.m, E);
  k_node_m<<<cdiv(C.n, 256), 256, 0, C.s>>>(C.asrc, C.adst, (unsigned*)C.m, C.wself, C.denom, C.n);
  k_fill_f32<<<cdiv((long long)C.n * 64, 256), 256, 0, C.s>>>(C.bnum, 0.0f, C.n * 64);
  k_edge_w<<<cdiv(E, 256), 256, 0, C.s>>>(row, col, C.asrc, C.adst, vmask, tmask, C.m, C.wbuf, C.denom, E);
  k_edge_num<<<cdiv((long long)E * 64, 256), 256, 0, C.s>>>(row, col, C.wbuf, C.bhg, C.bnum, E * 64);
  k_gat_final<<<cdiv((long long)C.n * 64, 256), 256, 0, C.s>>>(C.bnum, C.bhg, C.wself, C.denom, g.bias,
                                                               outb, C.n * 64, do_tanh);
}

static void run_lstm(const Ctx& C, const LstmP& l, const float* inA, const float* inB, int K)
{
  gemm(C, inA, inB, 64, K, l.wih, l.bih, C.bg, 256, 0);
  gemm(C, C.bh, nullptr, 64, 64, l.whh, l.bhh, C.bg, 256, 1);   // accumulate
  k_lstm<<<cdiv((long long)C.n * 64, 256), 256, 0, C.s>>>(C.bg, C.bh, C.bc, C.n * 64);
}

static void run_broadcast(const Ctx& C, const BroP& b, int rootOff)
{
  k_fill_u32<<<cdiv(C.n, 256), 256, 0, C.s>>>((unsigned*)C.maskA, 0u, C.n);
  k_set_roots<<<1, 256, 0, C.s>>>(C.maskA, C.roots, rootOff, 2, 256);
  k_init_x<<<cdiv((long long)C.n * 64, 256), 256, 0, C.s>>>(C.bx, C.feat, C.maskA, C.n * 64);
  k_fill_f32<<<cdiv((long long)C.n * 64, 256), 256, 0, C.s>>>(C.bh, 0.0f, C.n * 64);
  k_fill_f32<<<cdiv((long long)C.n * 64, 256), 256, 0, C.s>>>(C.bc, 0.0f, C.n * 64);
  int* cur = C.maskA; int* nxt = C.maskB;
  for (int layer = 0; layer < 2; ++layer) {
    // frontier (touched) computed from current mask BEFORE the GAT; using it as
    // the target mask is exact: non-frontier nodes reduce to self-only softmax.
    k_fill_u32<<<cdiv(C.n, 256), 256, 0, C.s>>>((unsigned*)nxt, 0u, C.n);
    k_touched<<<cdiv(C.Ebi, 256), 256, 0, C.s>>>(C.brow, C.bcol, cur, nxt, C.Ebi);
    run_gat(C, b.gat[layer], C.bx, C.brow, C.bcol, C.Ebi, cur, nxt, C.bt1, 1);
    run_lstm(C, b.lstm[layer], C.bt1, nullptr, 64);
    k_update_x<<<cdiv((long long)C.n * 64, 256), 256, 0, C.s>>>(C.bx, C.bh, C.feat, nxt, C.n * 64);
    int* t = cur; cur = nxt; nxt = t;
  }
  // result (send embedding) is left in C.bx
}

static void run_agg(const Ctx& C, const AggP& a, const float* send, int rootOff, int colOff)
{
  k_fill_u32<<<cdiv(C.n, 256), 256, 0, C.s>>>((unsigned*)C.tmask, 0u, C.n);
  k_set_roots<<<1, 256, 0, C.s>>>(C.tmask, C.roots, rootOff, 2, 256);
  gemm(C, C.feat, send, 64, 128, a.lin1_w, a.lin1_b, C.bx0, 64, 0);
  // GAT outputs only consumed at the 256 source nodes -> restrict scatter there
  run_gat(C, a.gat[0], C.bx0, C.erow, C.ecol, C.E, nullptr, C.tmask, C.bt1, 1);
  run_gat(C, a.gat[1], C.bx0, C.erow, C.ecol, C.E, nullptr, C.tmask, C.bt2, 1);
  k_fill_f32<<<cdiv((long long)C.n * 64, 256), 256, 0, C.s>>>(C.bh, 0.0f, C.n * 64);
  k_fill_f32<<<cdiv((long long)C.n * 64, 256), 256, 0, C.s>>>(C.bc, 0.0f, C.n * 64);
  run_lstm(C, a.lstm[0], C.bt1, C.bx0, 128);   // input = [ht1, x0]
  run_lstm(C, a.lstm[1], C.bt2, C.bh, 128);    // input = [ht2, h]   (h read before update)
  gemm(C, C.bh, nullptr, 64, 64, a.lin2_w, a.lin2_b, C.bhg, 64, 0);
  k_gather<<<cdiv(256 * 64, 256), 256, 0, C.s>>>(C.out, C.bhg, C.roots, rootOff, 2, colOff, 256);
}

extern "C" void kernel_launch(void* const* d_in, const int* in_sizes, int n_in,
                              void* d_out, int out_size, void* d_ws, size_t ws_size,
                              hipStream_t stream)
{
  (void)n_in; (void)out_size; (void)ws_size;
  Ctx C{};
  C.s = stream;
  C.feat  = (const float*)d_in[0];
  const int* eidx = (const int*)d_in[1];
  const int* bidx = (const int*)d_in[2];
  C.roots = (const int*)d_in[3];
  C.n   = in_sizes[0] / 64;
  C.E   = in_sizes[1] / 2;
  C.Ebi = in_sizes[2] / 2;
  C.erow = eidx;  C.ecol = eidx + C.E;
  C.brow = bidx;  C.bcol = bidx + C.Ebi;

  // params: top level in setup_inputs() insertion order; nested dicts in
  // jax tree-flatten (sorted-key) order.
  int pi = 4;
  AggP a_src = take_agg(d_in, pi);
  AggP a_tgt = take_agg(d_in, pi);
  BroP b_src = take_bro(d_in, pi);
  BroP b_tgt = take_bro(d_in, pi);

  // carve workspace
  char* p = (char*)d_ws;
  auto alloc = [&](size_t bytes) -> char* {
    char* r = p; p += (bytes + 255) & ~(size_t)255; return r;
  };
  size_t nv = (size_t)C.n * 64 * sizeof(float);
  C.bx   = (float*)alloc(nv);
  C.bh   = (float*)alloc(nv);
  C.bc   = (float*)alloc(nv);
  C.bhg  = (float*)alloc(nv);
  C.bnum = (float*)alloc(nv);
  C.bt1  = (float*)alloc(nv);
  C.bt2  = (float*)alloc(nv);
  C.bx0  = (float*)alloc(nv);
  C.bg   = (float*)alloc((size_t)C.n * 256 * sizeof(float));
  C.m     = (float*)alloc((size_t)C.n * sizeof(float));
  C.wself = (float*)alloc((size_t)C.n * sizeof(float));
  C.denom = (float*)alloc((size_t)C.n * sizeof(float));
  C.asrc  = (float*)alloc((size_t)C.n * sizeof(float));
  C.adst  = (float*)alloc((size_t)C.n * sizeof(float));
  C.maskA = (int*)alloc((size_t)C.n * sizeof(int));
  C.maskB = (int*)alloc((size_t)C.n * sizeof(int));
  C.tmask = (int*)alloc((size_t)C.n * sizeof(int));
  int Emax = C.E > C.Ebi ? C.E : C.Ebi;
  C.wbuf = (float*)alloc((size_t)Emax * sizeof(float));
  C.out = (float*)d_out;

  // forward: out = concat([source_embd, target_embd], 1)
  run_broadcast(C, b_src, 0);          // sources = s -> send_s in C.bx
  run_agg(C, a_src, C.bx, 1, 64);      // target_embd = agg(a_src, send_s)[t] -> cols 64..127
  run_broadcast(C, b_tgt, 1);          // sources = t -> send_t in C.bx
  run_agg(C, a_tgt, C.bx, 0, 0);       // source_embd = agg(a_tgt, send_t)[s] -> cols 0..63
}
